// AttentiveTransformer_90898687853205
// MI455X (gfx1250) — matrix operand order repaired and checked by reading, not compiled, not run on hardware
//
#include <hip/hip_runtime.h>

typedef _Float16 v16h __attribute__((ext_vector_type(16)));
typedef _Float16 v8h  __attribute__((ext_vector_type(8)));
typedef _Float16 v4h  __attribute__((ext_vector_type(4)));
typedef float    v8f  __attribute__((ext_vector_type(8)));
typedef float    v4f  __attribute__((ext_vector_type(4)));

#define H_DIM 256
#define F_DIM 512
#define TILE_M 32
#define LDS_A_STRIDE 264   // halves: 256 + 8 pad (bank-conflict avoidance)
#define LDS_H_STRIDE 516   // floats: 512 + 4 pad

__global__ void AttentiveTransformer_cvtW_kernel(const float* __restrict__ W,
                                                 _Float16* __restrict__ Wh, int n) {
  int i = blockIdx.x * blockDim.x + threadIdx.x;
  if (i < n) Wh[i] = (_Float16)W[i];
}

__device__ __forceinline__ float wred_sum(float v) {
  #pragma unroll
  for (int o = 16; o > 0; o >>= 1) v += __shfl_xor(v, o, 32);
  return v;
}
__device__ __forceinline__ float wred_max(float v) {
  #pragma unroll
  for (int o = 16; o > 0; o >>= 1) v = fmaxf(v, __shfl_xor(v, o, 32));
  return v;
}

__global__ __launch_bounds__(256)
void AttentiveTransformer_90898687853205_kernel(
    const float* __restrict__ x, const float* __restrict__ prior,
    const _Float16* __restrict__ Wh, const float* __restrict__ bias,
    const float* __restrict__ gamma, const float* __restrict__ beta,
    float* __restrict__ out)
{
  extern __shared__ char lds_raw[];
  float*    hT = (float*)lds_raw;                                    // 32 x 516 f32
  _Float16* A  = (_Float16*)(lds_raw + TILE_M * LDS_H_STRIDE * sizeof(float)); // 32 x 264 f16

  const int tid     = threadIdx.x;
  const int wave    = tid >> 5;
  const int lane    = tid & 31;
  const int half    = lane >> 4;      // which 16-lane half of the wave
  const int l16     = lane & 15;
  const int rowBase = blockIdx.x * TILE_M;

  // ---- Stage x tile (32 x 256 f32) -> LDS as f16, coalesced float4 loads ----
  {
    const v4f* x4 = (const v4f*)(x + (size_t)rowBase * H_DIM);
    #pragma unroll
    for (int i = 0; i < 8; ++i) {
      int idx = tid + 256 * i;            // 0..2047 float4s
      int r   = idx >> 6;                 // row 0..31
      int c4  = (idx & 63) << 2;          // col 0..252
      v4f v = x4[idx];
      v4h h; h.x = (_Float16)v.x; h.y = (_Float16)v.y;
             h.z = (_Float16)v.z; h.w = (_Float16)v.w;
      *(v4h*)&A[r * LDS_A_STRIDE + c4] = h;
    }
  }
  __syncthreads();

  // ---- GEMM: wave computes rows 0..31 x cols [wave*64, wave*64+64) ----
  v8f acc[2][4];
  #pragma unroll
  for (int rt = 0; rt < 2; ++rt)
    #pragma unroll
    for (int ct = 0; ct < 4; ++ct) { v8f z = {}; acc[rt][ct] = z; }

  const int colBase = wave * 64;

  #pragma unroll
  for (int kt = 0; kt < 8; ++kt) {
    const int k0 = kt * 32;
    // A fragments: 16-bit A 16x32 layout -> lane holds row (l16),
    // K = half*8 + [0..7] in v0..3 and half*8 + 16 + [0..7] in v4..7
    v16h af[2];
    #pragma unroll
    for (int rt = 0; rt < 2; ++rt) {
      const _Float16* ap = &A[(rt * 16 + l16) * LDS_A_STRIDE + k0 + half * 8];
      union { v16h v; v8h h[2]; } u;
      u.h[0] = *(const v8h*)ap;
      u.h[1] = *(const v8h*)(ap + 16);
      af[rt] = u.v;
    }
    // B fragments: 16-bit B 32x16 layout -> lane holds col (l16),
    // K = half*8 + [0..7] in v0..3 and half*8 + 16 + [0..7] in v4..7 (matches A)
    v16h bf[4];
    #pragma unroll
    for (int ct = 0; ct < 4; ++ct) {
      const _Float16* bp = Wh + (size_t)(colBase + ct * 16 + l16) * H_DIM + k0 + half * 8;
      union { v16h v; v8h h[2]; } u;
      u.h[0] = *(const v8h*)bp;
      u.h[1] = *(const v8h*)(bp + 16);
      bf[ct] = u.v;
    }
    #pragma unroll
    for (int rt = 0; rt < 2; ++rt)
      #pragma unroll
      for (int ct = 0; ct < 4; ++ct)
        acc[rt][ct] = __builtin_amdgcn_wmma_f32_16x16x32_f16(
            false, af[rt], false, bf[ct], (short)0, acc[rt][ct], false, false);
  }

  // ---- Spill accumulators to LDS h-tile (C layout: vgpr v -> M = v + 8*half) ----
  #pragma unroll
  for (int rt = 0; rt < 2; ++rt)
    #pragma unroll
    for (int ct = 0; ct < 4; ++ct) {
      const int col = colBase + ct * 16 + l16;
      #pragma unroll
      for (int v = 0; v < 8; ++v) {
        const int m = rt * 16 + v + half * 8;
        hT[m * LDS_H_STRIDE + col] = acc[rt][ct][v];
      }
    }
  __syncthreads();

  // ---- Fused epilogue: one row per wave, 4 rows each ----
  // Per-lane column set f = lane + 32*j (coalesced). Hoist per-column constants.
  float bia[16], gam[16], bet[16];
  #pragma unroll
  for (int j = 0; j < 16; ++j) {
    const int f = lane + 32 * j;
    bia[j] = bias[f]; gam[j] = gamma[f]; bet[j] = beta[f];
  }

  for (int p = 0; p < 4; ++p) {
    const int r = wave + 8 * p;
    const size_t grow = (size_t)(rowBase + r) * F_DIM;

    float z[16];
    float s1 = 0.f;
    #pragma unroll
    for (int j = 0; j < 16; ++j) {
      float hv = hT[r * LDS_H_STRIDE + lane + 32 * j] + bia[j];
      z[j] = hv;
      s1 += hv;
    }
    s1 = wred_sum(s1);
    const float mean = s1 * (1.0f / (float)F_DIM);
    float s2 = 0.f;
    #pragma unroll
    for (int j = 0; j < 16; ++j) { float d = z[j] - mean; s2 += d * d; }
    s2 = wred_sum(s2);
    const float rstd = rsqrtf(s2 * (1.0f / (float)F_DIM) + 1e-5f);

    #pragma unroll
    for (int j = 0; j < 16; ++j) {
      float nv = (z[j] - mean) * rstd * gam[j] + bet[j];
      z[j] = nv * prior[grow + lane + 32 * j];
    }

    // sparsemax: shift by max, bisect tau in [-1, 0] s.t. sum(max(z-tau,0)) = 1
    float m = z[0];
    #pragma unroll
    for (int j = 1; j < 16; ++j) m = fmaxf(m, z[j]);
    m = wred_max(m);
    #pragma unroll
    for (int j = 0; j < 16; ++j) z[j] -= m;

    float lo = -1.0f, hi = 0.0f;
    for (int it = 0; it < 30; ++it) {
      const float t = 0.5f * (lo + hi);
      float s = 0.f;
      #pragma unroll
      for (int j = 0; j < 16; ++j) s += fmaxf(z[j] - t, 0.f);
      s = wred_sum(s);             // wave-uniform
      lo = (s >= 1.0f) ? t : lo;
      hi = (s >= 1.0f) ? hi : t;
    }
    // exact tau from the (now fixed) support set
    const float t = 0.5f * (lo + hi);
    float cnt = 0.f, sum = 0.f;
    #pragma unroll
    for (int j = 0; j < 16; ++j) {
      if (z[j] > t) { cnt += 1.0f; sum += z[j]; }
    }
    cnt = wred_sum(cnt);
    sum = wred_sum(sum);
    const float tau = (sum - 1.0f) / fmaxf(cnt, 1.0f);

    #pragma unroll
    for (int j = 0; j < 16; ++j)
      out[grow + lane + 32 * j] = fmaxf(z[j] - tau, 0.f);
  }
}

extern "C" void kernel_launch(void* const* d_in, const int* in_sizes, int n_in,
                              void* d_out, int out_size, void* d_ws, size_t ws_size,
                              hipStream_t stream) {
  const float* x     = (const float*)d_in[0];
  const float* prior = (const float*)d_in[1];
  const float* W     = (const float*)d_in[2];
  const float* bias  = (const float*)d_in[3];
  const float* gamma = (const float*)d_in[4];
  const float* beta  = (const float*)d_in[5];
  float* out = (float*)d_out;

  _Float16* Wh = (_Float16*)d_ws;           // 512*256 f16 = 256 KB scratch
  const int nW = in_sizes[2];               // F*H = 131072
  const int B  = in_sizes[0] / H_DIM;       // 131072

  hipLaunchKernelGGL(AttentiveTransformer_cvtW_kernel,
                     dim3((nW + 255) / 256), dim3(256), 0, stream, W, Wh, nW);

  const size_t lds = (size_t)TILE_M * LDS_H_STRIDE * sizeof(float)
                   + (size_t)TILE_M * LDS_A_STRIDE * sizeof(_Float16);
  hipLaunchKernelGGL(AttentiveTransformer_90898687853205_kernel,
                     dim3(B / TILE_M), dim3(256), lds, stream,
                     x, prior, Wh, bias, gamma, beta, out);
}
